// VQEmbeddingEMA_54554674594282
// MI455X (gfx1250) — matrix operand order repaired and checked by reading, not compiled, not run on hardware
//
#include <hip/hip_runtime.h>
#include <float.h>
#include <math.h>

// ---------------------------------------------------------------------------
// VQ-VAE EMA vector quantization for MI455X (gfx1250, wave32, WMMA).
// Distance GEMM on V_WMMA_F32_16X16X32_BF16 using error-compensated bf16x3
// splitting (x·e ≈ xh·eh + xh·el + xl·eh, ~1e-5 rel err — argmin-safe),
// instead of the K=4 f32 WMMA which would leave the kernel compute-bound.
// Embeddings staged through LDS pre-split into packed bf16 hi/lo planes.
// ---------------------------------------------------------------------------

typedef __attribute__((ext_vector_type(16))) __bf16          v16bf;
typedef __attribute__((ext_vector_type(16))) unsigned short  v16u;
typedef __attribute__((ext_vector_type(8)))  float           v8f;

#define N_CB    4
#define M_EMB   512
#define D_EMB   32
#define L_LAT   64
#define B_BATCH 1024
#define T_POS   (B_BATCH * L_LAT)      // 65536 positions per codebook
#define DECAYF  0.999f
#define CCOST   0.25f
#define EPSF    1e-5f

// Output layout (floats, concatenated in reference tuple order)
#define OUT_ZQ   0                      // [B, N*D*L] = 8388608
#define OUT_LOSS 8388608                // scalar
#define OUT_PERP 8388609                // scalar
#define OUT_IDX  8388610                // [B, N, L] = 262144 (indices as float)
#define OUT_EMB  8650754                // [N, M, D] = 65536
#define OUT_CNT  8716290                // [N, M]    = 2048
#define OUT_W    8718338                // [N, M, D] = 65536

// Workspace layout (floats)
#define WS_COUNTS 0                     // [N, M]    = 2048
#define WS_DW     2048                  // [N, M, D] = 65536
#define WS_LOSS   67584                 // scalar accumulator
#define WS_PERP   67585                 // [N] per-codebook perplexity
#define WS_TOTAL  67589

// round-to-nearest-even f32 -> bf16 (bit pattern)
static __device__ __forceinline__ unsigned short f2bf(float f) {
    unsigned u = __float_as_uint(f);
    u += 0x7FFFu + ((u >> 16) & 1u);
    return (unsigned short)(u >> 16);
}
static __device__ __forceinline__ float bf2f(unsigned short h) {
    return __uint_as_float((unsigned)h << 16);
}

union V16 { v16u u; v16bf v; uint4 q[2]; };

// ---------------------------------------------------------------------------
__global__ void vq_zero(float* __restrict__ ws) {
    int i = blockIdx.x * 256 + threadIdx.x;
    if (i < WS_TOTAL) ws[i] = 0.0f;
}

// ---------------------------------------------------------------------------
// Main kernel: 8 waves / block, each wave owns one 16-position tile.
// grid = (T/16/8, N) = (512, 4), block = 256.
// ---------------------------------------------------------------------------
__global__ __launch_bounds__(256)
void vq_main(const float* __restrict__ x, const float* __restrict__ emb,
             float* __restrict__ out, float* __restrict__ ws) {
    __shared__ unsigned short embHiS[256 * 32];   // 16 KB: bf16(e)
    __shared__ unsigned short embLoS[256 * 32];   // 16 KB: bf16(e - hi)
    __shared__ float          esqS[256];          // exact f32 ||e||^2
    __shared__ int            idxS[8 * 16];

    const int n    = blockIdx.y;
    const int wave = threadIdx.x >> 5;
    const int ln   = threadIdx.x & 31;
    const int hi   = ln >> 4;           // half-wave id
    const int row  = ln & 15;           // A row / D sub-row / B-N column

    const int tile = blockIdx.x * 8 + wave;
    const int t0   = tile * 16;
    const int b    = t0 >> 6;           // t = b*L + l, L = 64
    const int l0   = t0 & 63;

    // x_flat[n][t = b*L+l][d] lives at x[b*8192 + n*2048 + d*64 + l]
    const float* xb = x + (size_t)b * 8192 + n * 2048 + l0;
    __builtin_prefetch(xb, 0, 1);       // global_prefetch_b8

    // --- A matrix (16x32 bf16) split hi/lo. Contiguous K order (matches B):
    // lane holds row = ln%16; element j -> K = j + 16*(ln/16)
    V16 AH, AL;
#pragma unroll
    for (int j = 0; j < 16; ++j) {
        const int K = j + hi * 16;
        const float xv = xb[K * 64 + row];
        const unsigned short h = f2bf(xv);
        AH.u[j] = h;
        AL.u[j] = f2bf(xv - bf2f(h));
    }

    float minv[8];
    int   mini[8];
#pragma unroll
    for (int r = 0; r < 8; ++r) { minv[r] = FLT_MAX; mini[r] = 0; }

    const float* embG = emb + (size_t)n * (M_EMB * D_EMB);

    for (int half = 0; half < 2; ++half) {
        // Stage 256 embedding rows: split into bf16 hi/lo planes + f32 e_sq.
        {
            const int m = threadIdx.x;   // 256 threads, one row each
            const float4* er = (const float4*)(embG + (size_t)(half * 256 + m) * 32);
            unsigned* hS = (unsigned*)embHiS + m * 16;
            unsigned* lS = (unsigned*)embLoS + m * 16;
            float s = 0.0f;
#pragma unroll
            for (int i = 0; i < 8; ++i) {
                const float4 v = er[i];
                s += v.x * v.x + v.y * v.y + v.z * v.z + v.w * v.w;
                const unsigned short h0 = f2bf(v.x), h1 = f2bf(v.y);
                const unsigned short h2 = f2bf(v.z), h3 = f2bf(v.w);
                hS[i * 2 + 0] = (unsigned)h0 | ((unsigned)h1 << 16);
                hS[i * 2 + 1] = (unsigned)h2 | ((unsigned)h3 << 16);
                const unsigned short l0s = f2bf(v.x - bf2f(h0));
                const unsigned short l1s = f2bf(v.y - bf2f(h1));
                const unsigned short l2s = f2bf(v.z - bf2f(h2));
                const unsigned short l3s = f2bf(v.w - bf2f(h3));
                lS[i * 2 + 0] = (unsigned)l0s | ((unsigned)l1s << 16);
                lS[i * 2 + 1] = (unsigned)l2s | ((unsigned)l3s << 16);
            }
            esqS[m] = s;
        }
        __syncthreads();

        // 16 m-tiles of 16 columns in this half; 3 bf16 WMMAs per tile.
        for (int mt = 0; mt < 16; ++mt) {
            const float esq = esqS[mt * 16 + row];
            // B (32x16 bf16): lane = column (ln%16); lanes 0-15 hold K=0-15,
            // lanes 16-31 hold K=16-31 -> one contiguous 32B chunk of the row.
            const int rbase = (mt * 16 + row) * 32 + hi * 16;   // ushort offset
            V16 BH, BL;
            BH.q[0] = *(const uint4*)(embHiS + rbase + 0);
            BH.q[1] = *(const uint4*)(embHiS + rbase + 8);
            BL.q[0] = *(const uint4*)(embLoS + rbase + 0);
            BL.q[1] = *(const uint4*)(embLoS + rbase + 8);

            v8f acc = {0.f, 0.f, 0.f, 0.f, 0.f, 0.f, 0.f, 0.f};
            acc = __builtin_amdgcn_wmma_f32_16x16x32_bf16(
                    false, AH.v, false, BH.v, (short)0, acc, false, false);
            acc = __builtin_amdgcn_wmma_f32_16x16x32_bf16(
                    false, AH.v, false, BL.v, (short)0, acc, false, false);
            acc = __builtin_amdgcn_wmma_f32_16x16x32_bf16(
                    false, AL.v, false, BH.v, (short)0, acc, false, false);

            const int mbase = half * 256 + mt * 16 + row;
#pragma unroll
            for (int r = 0; r < 8; ++r) {
                // distance score (x^2 term is row-constant, irrelevant for argmin)
                float score = esq - 2.0f * acc[r];
                if (score < minv[r]) { minv[r] = score; mini[r] = mbase; }
            }
        }
        __syncthreads();   // before LDS is re-staged with the next half
    }

    // Cross-lane argmin within each 16-lane half (lanes share the same rows).
#pragma unroll
    for (int mask = 1; mask < 16; mask <<= 1) {
#pragma unroll
        for (int r = 0; r < 8; ++r) {
            float ov = __shfl_xor(minv[r], mask, 32);
            int   oi = __shfl_xor(mini[r], mask, 32);
            if (ov < minv[r] || (ov == minv[r] && oi < mini[r])) {
                minv[r] = ov; mini[r] = oi;
            }
        }
    }
    if (row == 0) {
#pragma unroll
        for (int r = 0; r < 8; ++r) idxS[wave * 16 + hi * 8 + r] = mini[r];
    }
    __syncthreads();

    // Emit indices (as float) + histogram counts.
    if (ln < 16) {
        const int id = idxS[wave * 16 + ln];
        out[OUT_IDX + (size_t)b * (N_CB * L_LAT) + n * L_LAT + l0 + ln] = (float)id;
        atomicAdd(&ws[WS_COUNTS + n * M_EMB + id], 1.0f);
    }

    // z_q gather + dw scatter + commitment-loss partial, one coalesced pass.
    const int    myIdx = idxS[wave * 16 + row];
    const float* qrow  = embG + (size_t)myIdx * 32;
    float lsum = 0.0f;
#pragma unroll
    for (int d0 = 0; d0 < 16; ++d0) {
        const int d = d0 * 2 + hi;
        const float xv = xb[d * 64 + row];
        const float q  = qrow[d];
        out[OUT_ZQ + (size_t)b * 8192 + n * 2048 + d * 64 + l0 + row] = q;
        atomicAdd(&ws[WS_DW + ((size_t)n * M_EMB + myIdx) * 32 + d], xv);
        const float df = xv - q;
        lsum += df * df;
    }
#pragma unroll
    for (int mask = 1; mask < 32; mask <<= 1) lsum += __shfl_xor(lsum, mask, 32);
    if (ln == 0) atomicAdd(&ws[WS_LOSS], lsum);
}

// ---------------------------------------------------------------------------
// EMA / perplexity finalize: one block per codebook, one thread per embedding.
// ---------------------------------------------------------------------------
__global__ __launch_bounds__(512)
void vq_finalize(const float* __restrict__ ema_count,
                 const float* __restrict__ ema_weight,
                 float* __restrict__ out, float* __restrict__ ws) {
    __shared__ float red[512];
    const int n = blockIdx.x, tid = threadIdx.x;

    const float cnt = ws[WS_COUNTS + n * M_EMB + tid];
    const float raw = DECAYF * ema_count[n * M_EMB + tid] + (1.0f - DECAYF) * cnt;

    red[tid] = raw;
    __syncthreads();
    for (int s = 256; s > 0; s >>= 1) {
        if (tid < s) red[tid] += red[tid + s];
        __syncthreads();
    }
    const float ntot = red[0];
    __syncthreads();

    const float nec = (raw + EPSF) / (ntot + (float)M_EMB * EPSF) * ntot;
    out[OUT_CNT + n * M_EMB + tid] = nec;

    // perplexity per codebook
    const float p = cnt / (float)T_POS;
    red[tid] = p * logf(p + 1e-10f);
    __syncthreads();
    for (int s = 256; s > 0; s >>= 1) {
        if (tid < s) red[tid] += red[tid + s];
        __syncthreads();
    }
    if (tid == 0) ws[WS_PERP + n] = expf(-red[0]);

    // EMA weight + new embedding for this thread's row
    const float invc = 1.0f / nec;
#pragma unroll
    for (int d = 0; d < 32; ++d) {
        const size_t e = ((size_t)n * M_EMB + tid) * 32 + d;
        const float w = DECAYF * ema_weight[e] + (1.0f - DECAYF) * ws[WS_DW + e];
        out[OUT_W + e]   = w;
        out[OUT_EMB + e] = w * invc;
    }
}

// ---------------------------------------------------------------------------
__global__ void vq_scalars(float* __restrict__ out, const float* __restrict__ ws) {
    out[OUT_LOSS] = CCOST * ws[WS_LOSS] / (float)(N_CB * B_BATCH * L_LAT * D_EMB);
    out[OUT_PERP] = ws[WS_PERP + 0] + ws[WS_PERP + 1] + ws[WS_PERP + 2] + ws[WS_PERP + 3];
}

// ---------------------------------------------------------------------------
extern "C" void kernel_launch(void* const* d_in, const int* in_sizes, int n_in,
                              void* d_out, int out_size, void* d_ws, size_t ws_size,
                              hipStream_t stream) {
    const float* x          = (const float*)d_in[0];
    const float* embedding  = (const float*)d_in[1];
    const float* ema_count  = (const float*)d_in[2];
    const float* ema_weight = (const float*)d_in[3];
    float* out = (float*)d_out;
    float* ws  = (float*)d_ws;

    vq_zero<<<(WS_TOTAL + 255) / 256, 256, 0, stream>>>(ws);
    vq_main<<<dim3(T_POS / 16 / 8, N_CB), 256, 0, stream>>>(x, embedding, out, ws);
    vq_finalize<<<N_CB, 512, 0, stream>>>(ema_count, ema_weight, out, ws);
    vq_scalars<<<1, 1, 0, stream>>>(out, ws);
}
